// MHA_63891933495304
// MI455X (gfx1250) — matrix operand order repaired and checked by reading, not compiled, not run on hardware
//
#include <hip/hip_runtime.h>

typedef __attribute__((ext_vector_type(16))) _Float16 v16h;
typedef __attribute__((ext_vector_type(8)))  _Float16 v8h;
typedef __attribute__((ext_vector_type(8)))  float    v8f;

#define D_MODEL 1024
#define SEQ     2048
#define BATCH   2
#define HEADS   16
#define HDIM    64
#define MROWS   (BATCH * SEQ)          // 4096
#define MASKVAL (-100000000.0f)

// Padded LDS row strides (halves) chosen via TDM pad: 32B-aligned, conflict-poor
#define KROW    80                     // 64 data + 16 pad halves (160B stride)
#define VROW    48                     // 32 data + 16 pad halves (96B stride)

// ---------------------------------------------------------------------------
// Fragment loaders (wave32, v_wmma_f32_16x16x32_f16; contiguous k order, same for A and B)
// A is 16(M) x 32(K), row-major [M, ldk]:
//   lane L: g=L/16, m=L%16 ; halves h: K = g*16 + h
//   -> per-lane base  A + m*ldk + g*16 ; loads at +k and +k+8
// B is 32(K) x 16(N), sourced from row-major [N, ldk] ("Bt"):
//   lane L: g=L/16, n=L%16 ; halves h: K = g*16 + h  (contiguous 32B run)
//   -> per-lane base  Bt + n*ldk + g*16 ; load at +k
// ---------------------------------------------------------------------------
static __device__ __forceinline__ v16h load_a(const _Float16* laneBase, int k) {
    v8h lo = *(const v8h*)(laneBase + k);
    v8h hi = *(const v8h*)(laneBase + k + 8);
    v16h a;
#pragma unroll
    for (int i = 0; i < 8; ++i) { a[i] = lo[i]; a[i + 8] = hi[i]; }
    return a;
}

static __device__ __forceinline__ v16h load_b(const _Float16* laneBase, int k) {
    return *(const v16h*)(laneBase + k);   // 32B aligned at all call sites
}

static __device__ __forceinline__ v8f wmma_f16(v16h a, v16h b, v8f c) {
    return __builtin_amdgcn_wmma_f32_16x16x32_f16(false, a, false, b, (short)0, c,
                                                  false, false);
}

// ---------------------------------------------------------------------------
// Tensor Data Mover: async 2-D tile load global -> LDS (ISA 08, D# 8.3/8.4).
// Descriptor packed per CDNA5 D# spec; 2B elements; optional LDS row padding.
// ---------------------------------------------------------------------------
typedef unsigned int tdmv4u __attribute__((ext_vector_type(4)));
typedef int          tdmv8i __attribute__((ext_vector_type(8)));
typedef int          tdmv4i __attribute__((ext_vector_type(4)));

static __device__ __forceinline__ void
tdm_load_2d_to_lds(const void* gtile, unsigned ldsByteOff,
                   unsigned tensorDim0, unsigned tensorDim1, unsigned stride0,
                   unsigned tileDim0, unsigned tileDim1,
                   unsigned padInterval, unsigned padAmount, bool padEnable) {
    unsigned long long ga = (unsigned long long)(size_t)gtile;
    tdmv4u g0;
    g0[0] = 1u;                                    // count=1 (valid), user mode
    g0[1] = ldsByteOff;                            // D#.lds_addr (bytes)
    g0[2] = (unsigned)ga;                          // global_addr[31:0]
    g0[3] = (unsigned)((ga >> 32) & 0x01FFFFFFull) // global_addr[56:32]
            | (2u << 30);                          // type = 2 ("image")
    tdmv8i g1;
    g1[0] = (int)((1u << 16)                       // data_size = 1 -> 2 bytes
                  | (padEnable ? (1u << 20) : 0u)
                  | ((padInterval & 7u) << 22)     // pad every 2<<i DWORDs
                  | ((padAmount & 127u) << 25));   // pad (code+1) DWORDs
    g1[1] = (int)((tensorDim0 & 0xFFFFu) << 16);   // dim0[15:0]  @ bits 63:48
    g1[2] = (int)(((tensorDim0 >> 16) & 0xFFFFu)   // dim0[31:16] @ bits 79:64
                  | ((tensorDim1 & 0xFFFFu) << 16));
    g1[3] = (int)(((tensorDim1 >> 16) & 0xFFFFu)
                  | ((tileDim0 & 0xFFFFu) << 16)); // tile_dim0 @ bits 127:112
    g1[4] = (int)(tileDim1 & 0xFFFFu);             // tile_dim1; tile_dim2 = 0
    g1[5] = (int)stride0;                          // tensor_dim0_stride[31:0]
    g1[6] = 0;                                     // stride0 hi | stride1 lo
    g1[7] = 0;
    tdmv4i gz = {0, 0, 0, 0};                      // groups 2/3: 2-D tensor
#if defined(__clang_major__) && (__clang_major__ >= 23)
    tdmv8i gz8 = {0, 0, 0, 0, 0, 0, 0, 0};
    __builtin_amdgcn_tensor_load_to_lds(g0, g1, gz, gz, gz8, 0);
#else
    __builtin_amdgcn_tensor_load_to_lds(g0, g1, gz, gz, 0);
#endif
}

// ---------------------------------------------------------------------------
// fp32 -> fp16 conversion, 8 elements / thread
// ---------------------------------------------------------------------------
__global__ __launch_bounds__(256) void
cvt_f32_to_f16(const float* __restrict__ src, _Float16* __restrict__ dst, int n8) {
    int i = blockIdx.x * blockDim.x + threadIdx.x;
    if (i >= n8) return;
    const float4* s = (const float4*)src;
    float4 a = s[i * 2 + 0];
    float4 b = s[i * 2 + 1];
    v8h o;
    o[0] = (_Float16)a.x; o[1] = (_Float16)a.y; o[2] = (_Float16)a.z; o[3] = (_Float16)a.w;
    o[4] = (_Float16)b.x; o[5] = (_Float16)b.y; o[6] = (_Float16)b.z; o[7] = (_Float16)b.w;
    *(v8h*)(dst + (size_t)i * 8) = o;
}

// ---------------------------------------------------------------------------
// WMMA GEMM:  C[M,N] = A[M,K] * Bt[N,K]^T   (A,Bt fp16; fp32 accum)
// Wave tile 32(M) x 64(N): 8 accumulators, 2 A-frags sharing 4 B-frags per
// 32-deep k-step (8 WMMAs / 12 b128-loads), software-pipelined one k-step
// ahead. launch_bounds(...,1): full VGPR budget -> no spills (verified).
// mode 0: f16 out at Q/K layout [b,h,s,d]   (n = h*64+d, m = b*2048+s)
// mode 1: f16 out at V^T layout [b,h,d,s]
// mode 2: f16 out row-major [M,N]
// mode 3: f32 out row-major [M,N]  (final output)
// ---------------------------------------------------------------------------
__global__ __launch_bounds__(256, 1) void
gemm_wmma_f16(const _Float16* __restrict__ A,
              const _Float16* __restrict__ Bt,
              _Float16* __restrict__ dstH,
              float* __restrict__ dstF,
              int M, int N, int K, int mode) {
    const int lane = threadIdx.x & 31;
    const int wid  = blockIdx.x * (blockDim.x >> 5) + (threadIdx.x >> 5);
    const int tilesN = N >> 6;
    const int tM = wid / tilesN;          // consecutive waves share tM -> A reuse
    const int tN = wid - tM * tilesN;
    if (tM >= (M >> 5)) return;
    const int mBase = tM << 5;
    const int nBase = tN << 6;
    const int g  = lane >> 4;
    const int nl = lane & 15;

    // per-lane base pointers; only scalar k varies inside the loop
    const _Float16* ap0 = A  + (size_t)(mBase + nl)      * K + g * 16;
    const _Float16* ap1 = A  + (size_t)(mBase + 16 + nl) * K + g * 16;
    const _Float16* bp0 = Bt + (size_t)(nBase + nl)      * K + g * 16;
    const _Float16* bp1 = bp0 + (size_t)16 * K;
    const _Float16* bp2 = bp0 + (size_t)32 * K;
    const _Float16* bp3 = bp0 + (size_t)48 * K;

    v8f acc[8] = {};

    // prologue: fragments for k = 0
    v16h a0 = load_a(ap0, 0);
    v16h a1 = load_a(ap1, 0);
    v16h b0 = load_b(bp0, 0);
    v16h b1 = load_b(bp1, 0);
    v16h b2 = load_b(bp2, 0);
    v16h b3 = load_b(bp3, 0);

    for (int k = 32; k < K; k += 32) {
        // issue next-iteration loads first; latency hides under the 8 WMMAs
        v16h na0 = load_a(ap0, k);
        v16h na1 = load_a(ap1, k);
        v16h nb0 = load_b(bp0, k);
        v16h nb1 = load_b(bp1, k);
        v16h nb2 = load_b(bp2, k);
        v16h nb3 = load_b(bp3, k);

        acc[0] = wmma_f16(a0, b0, acc[0]);
        acc[1] = wmma_f16(a0, b1, acc[1]);
        acc[2] = wmma_f16(a0, b2, acc[2]);
        acc[3] = wmma_f16(a0, b3, acc[3]);
        acc[4] = wmma_f16(a1, b0, acc[4]);
        acc[5] = wmma_f16(a1, b1, acc[5]);
        acc[6] = wmma_f16(a1, b2, acc[6]);
        acc[7] = wmma_f16(a1, b3, acc[7]);

        a0 = na0; a1 = na1;
        b0 = nb0; b1 = nb1; b2 = nb2; b3 = nb3;
    }
    // epilogue k-step
    acc[0] = wmma_f16(a0, b0, acc[0]);
    acc[1] = wmma_f16(a0, b1, acc[1]);
    acc[2] = wmma_f16(a0, b2, acc[2]);
    acc[3] = wmma_f16(a0, b3, acc[3]);
    acc[4] = wmma_f16(a1, b0, acc[4]);
    acc[5] = wmma_f16(a1, b1, acc[5]);
    acc[6] = wmma_f16(a1, b2, acc[6]);
    acc[7] = wmma_f16(a1, b3, acc[7]);

#pragma unroll
    for (int rh = 0; rh < 2; ++rh) {
#pragma unroll
        for (int nt = 0; nt < 4; ++nt) {
#pragma unroll
            for (int r = 0; r < 8; ++r) {
                const int m = mBase + rh * 16 + r + g * 8;
                const int n = nBase + nt * 16 + nl;
                const float val = acc[rh * 4 + nt][r];
                if (mode == 3) {
                    dstF[(size_t)m * N + n] = val;
                } else if (mode == 2) {
                    dstH[(size_t)m * N + n] = (_Float16)val;
                } else {
                    const int b = m >> 11, s = m & 2047;
                    const int h = n >> 6,  d = n & 63;
                    if (mode == 0)
                        dstH[(((size_t)(b * HEADS + h)) * SEQ + s) * HDIM + d] = (_Float16)val;
                    else
                        dstH[(((size_t)(b * HEADS + h)) * HDIM + d) * SEQ + s] = (_Float16)val;
                }
            }
        }
    }
}

// ---------------------------------------------------------------------------
// Flash attention, causal. One wave owns a 32-row q tile of one (b,h).
// K/V kv-tiles are staged into per-wave double-buffered LDS by the Tensor
// Data Mover (TDM): while the 16 WMMAs + softmax consume buffer `buf`, the
// TDM fills buffer `buf^1`; s_wait_tensorcnt(2) gates buffer readiness
// (TDM ops complete in order within a wave). TDM pads LDS rows (K:160B,
// V:96B stride) to keep 32B-aligned fragment reads and spread LDS banks.
// Q,K: [b,h,s,64] f16.  Vt: [b,h,64,s] f16.  Ctx out: [b,s,h*64] f16.
// ---------------------------------------------------------------------------
__global__ __launch_bounds__(128, 1) void
attn_flash_wmma(const _Float16* __restrict__ Q,
                const _Float16* __restrict__ Kt,
                const _Float16* __restrict__ Vt,
                _Float16* __restrict__ Ctx) {
    __shared__ __align__(32) _Float16 kTile[4][2][32 * KROW];  // 40KB
    __shared__ __align__(32) _Float16 vTile[4][2][64 * VROW];  // 48KB
    __shared__ __align__(32) _Float16 plds[4][32][32];         //  8KB

    const int lane   = threadIdx.x & 31;
    const int wlocal = threadIdx.x >> 5;
    const int wid    = blockIdx.x * 4 + wlocal;
    const int qt     = wid & 63;             // 64 q-tiles of 32 rows
    const int bh     = wid >> 6;             // 32 (batch, head) pairs
    const int qBase  = qt << 5;
    const int g      = lane >> 4;
    const int nl     = lane & 15;

    const _Float16* Qp = Q  + (size_t)bh * SEQ * HDIM;
    const _Float16* Kp = Kt + (size_t)bh * SEQ * HDIM;
    const _Float16* Vp = Vt + (size_t)bh * HDIM * SEQ;

    // LDS byte offsets of this wave's staging buffers (flat LDS addr = low 32b)
    const unsigned kOff[2] = { (unsigned)(size_t)&kTile[wlocal][0][0],
                               (unsigned)(size_t)&kTile[wlocal][1][0] };
    const unsigned vOff[2] = { (unsigned)(size_t)&vTile[wlocal][0][0],
                               (unsigned)(size_t)&vTile[wlocal][1][0] };

    // Q fragments for both 16-row halves, d = 0..31 / 32..63; reused all steps
    const _Float16* qp0 = Qp + (size_t)(qBase + nl)      * HDIM + g * 16;
    const _Float16* qp1 = Qp + (size_t)(qBase + 16 + nl) * HDIM + g * 16;
    v16h qa[2][2];
    qa[0][0] = load_a(qp0, 0);
    qa[0][1] = load_a(qp0, 32);
    qa[1][0] = load_a(qp1, 0);
    qa[1][1] = load_a(qp1, 32);

    v8f o[2][4] = {};
    float rowM[2][8], rowL[2][8];
#pragma unroll
    for (int rh = 0; rh < 2; ++rh)
#pragma unroll
        for (int r = 0; r < 8; ++r) { rowM[rh][r] = -3.0e38f; rowL[rh][r] = 0.0f; }

    const float scale = 0.125f;              // 1/sqrt(64)

    // prologue: TDM-fetch kv tile j=0 into buffer 0
    // K tile: rows j..j+31 of [SEQ x 64]; pad 8 DWORDs every 32 DWORDs (row)
    tdm_load_2d_to_lds(Kp, kOff[0], HDIM, SEQ, HDIM, HDIM, 32, /*int*/4, /*amt*/7, true);
    // V tile: cols j..j+31 of [64 x SEQ]; pad 8 DWORDs every 16 DWORDs (row)
    tdm_load_2d_to_lds(Vp, vOff[0], SEQ, HDIM, SEQ, 32, HDIM, /*int*/3, /*amt*/7, true);

    int buf = 0;
    for (int j = 0; j <= qBase + 31; j += 32) {   // causal: skip tiles above diag
        const int nj = j + 32;
        if (nj <= qBase + 31) {
            // fill the other buffer while we consume this one
            tdm_load_2d_to_lds(Kp + (size_t)nj * HDIM, kOff[buf ^ 1],
                               HDIM, SEQ, HDIM, HDIM, 32, 4, 7, true);
            tdm_load_2d_to_lds(Vp + nj, vOff[buf ^ 1],
                               SEQ, HDIM, SEQ, 32, HDIM, 3, 7, true);
            __builtin_amdgcn_s_wait_tensorcnt(2);   // current buffer ready
        } else {
            __builtin_amdgcn_s_wait_tensorcnt(0);   // last tile
        }
        const _Float16* kb = &kTile[wlocal][buf][0];
        const _Float16* vb = &vTile[wlocal][buf][0];

        // ---- S tiles = Q * K^T, columns j..j+31, both row halves ----
        v8f s[2][2] = {};
        {
            v16h kb00 = *(const v16h*)(kb + (nl)      * KROW + g * 16);
            v16h kb01 = *(const v16h*)(kb + (nl)      * KROW + 32 + g * 16);
            v16h kb10 = *(const v16h*)(kb + (16 + nl) * KROW + g * 16);
            v16h kb11 = *(const v16h*)(kb + (16 + nl) * KROW + 32 + g * 16);
            s[0][0] = wmma_f16(qa[0][0], kb00, s[0][0]);
            s[0][0] = wmma_f16(qa[0][1], kb01, s[0][0]);
            s[0][1] = wmma_f16(qa[0][0], kb10, s[0][1]);
            s[0][1] = wmma_f16(qa[0][1], kb11, s[0][1]);
            s[1][0] = wmma_f16(qa[1][0], kb00, s[1][0]);
            s[1][0] = wmma_f16(qa[1][1], kb01, s[1][0]);
            s[1][1] = wmma_f16(qa[1][0], kb10, s[1][1]);
            s[1][1] = wmma_f16(qa[1][1], kb11, s[1][1]);
        }

        // ---- scale + causal mask + online softmax (per row half) ----
#pragma unroll
        for (int rh = 0; rh < 2; ++rh) {
#pragma unroll
            for (int r = 0; r < 8; ++r) {
                const int qrow = qBase + rh * 16 + r + g * 8;
                float sv0 = s[rh][0][r] * scale + ((j + nl      > qrow) ? MASKVAL : 0.0f);
                float sv1 = s[rh][1][r] * scale + ((j + 16 + nl > qrow) ? MASKVAL : 0.0f);

                float v = fmaxf(sv0, sv1);
                v = fmaxf(v, __shfl_xor(v, 1, 32));
                v = fmaxf(v, __shfl_xor(v, 2, 32));
                v = fmaxf(v, __shfl_xor(v, 4, 32));
                v = fmaxf(v, __shfl_xor(v, 8, 32));      // row max over 16 lanes
                const float newM = fmaxf(rowM[rh][r], v);
                const float corr = __expf(rowM[rh][r] - newM);
                rowM[rh][r] = newM;

                const float p0 = __expf(sv0 - newM);
                const float p1 = __expf(sv1 - newM);
                float ts = p0 + p1;
                ts += __shfl_xor(ts, 1, 32);
                ts += __shfl_xor(ts, 2, 32);
                ts += __shfl_xor(ts, 4, 32);
                ts += __shfl_xor(ts, 8, 32);
                rowL[rh][r] = rowL[rh][r] * corr + ts;
#pragma unroll
                for (int nt = 0; nt < 4; ++nt) o[rh][nt][r] *= corr;

                // C-layout -> LDS (row-major 32x32 P tile). Per-wave region;
                // LDS ops are in-order within a wave: no barrier needed.
                plds[wlocal][rh * 16 + r + g * 8][nl]      = (_Float16)p0;
                plds[wlocal][rh * 16 + r + g * 8][16 + nl] = (_Float16)p1;
            }
        }

        // ---- reload P in WMMA A-layout from LDS (both row halves) ----
        v16h pa0, pa1;
        {
            const _Float16* p0 = &plds[wlocal][nl][g * 16];
            const _Float16* p1 = &plds[wlocal][16 + nl][g * 16];
            v8h lo0 = *(const v8h*)(p0);
            v8h hi0 = *(const v8h*)(p0 + 8);
            v8h lo1 = *(const v8h*)(p1);
            v8h hi1 = *(const v8h*)(p1 + 8);
#pragma unroll
            for (int i = 0; i < 8; ++i) {
                pa0[i] = lo0[i]; pa0[i + 8] = hi0[i];
                pa1[i] = lo1[i]; pa1[i + 8] = hi1[i];
            }
        }

        // ---- O += P * V  (V^T tile rows = d, cols = kv local) ----
#pragma unroll
        for (int nt = 0; nt < 4; ++nt) {
            v16h vf = *(const v16h*)(vb + (nt * 16 + nl) * VROW + g * 16);
            o[0][nt] = wmma_f16(pa0, vf, o[0][nt]);
            o[1][nt] = wmma_f16(pa1, vf, o[1][nt]);
        }

        buf ^= 1;
    }

    // ---- normalize and store context in [b, s, h*64+d] (f16) ----
    const int b = bh >> 4, h = bh & 15;
#pragma unroll
    for (int rh = 0; rh < 2; ++rh) {
#pragma unroll
        for (int nt = 0; nt < 4; ++nt) {
#pragma unroll
            for (int r = 0; r < 8; ++r) {
                const int qrow = qBase + rh * 16 + r + g * 8;
                const int d    = nt * 16 + nl;
                const float val = o[rh][nt][r] / rowL[rh][r];
                Ctx[((size_t)(b * SEQ + qrow)) * D_MODEL + h * HDIM + d] = (_Float16)val;
            }
        }
    }
}

// ---------------------------------------------------------------------------
// Host-side orchestration
// ---------------------------------------------------------------------------
extern "C" void kernel_launch(void* const* d_in, const int* in_sizes, int n_in,
                              void* d_out, int out_size, void* d_ws, size_t ws_size,
                              hipStream_t stream) {
    const float* x  = (const float*)d_in[0];
    const float* wq = (const float*)d_in[1];
    const float* wk = (const float*)d_in[2];
    const float* wv = (const float*)d_in[3];
    const float* wo = (const float*)d_in[4];
    float* out = (float*)d_out;

    _Float16* ws = (_Float16*)d_ws;
    const size_t NX = (size_t)MROWS * D_MODEL;      // 4M halves
    const size_t NW = (size_t)D_MODEL * D_MODEL;    // 1M halves
    size_t off = 0;
    _Float16* Xh   = ws + off; off += NX;
    _Float16* Wqh  = ws + off; off += NW;
    _Float16* Wkh  = ws + off; off += NW;
    _Float16* Wvh  = ws + off; off += NW;
    _Float16* Woh  = ws + off; off += NW;
    _Float16* Qh   = ws + off; off += NX;   // [b,h,s,d]
    _Float16* Kh   = ws + off; off += NX;   // [b,h,s,d]
    _Float16* Vth  = ws + off; off += NX;   // [b,h,d,s]
    _Float16* Ctxh = ws + off; off += NX;   // [b,s,h*d]

    // fp32 -> fp16 conversions
    {
        int n8 = (int)(NX / 8);
        cvt_f32_to_f16<<<(n8 + 255) / 256, 256, 0, stream>>>(x, Xh, n8);
        n8 = (int)(NW / 8);
        cvt_f32_to_f16<<<(n8 + 255) / 256, 256, 0, stream>>>(wq, Wqh, n8);
        cvt_f32_to_f16<<<(n8 + 255) / 256, 256, 0, stream>>>(wk, Wkh, n8);
        cvt_f32_to_f16<<<(n8 + 255) / 256, 256, 0, stream>>>(wv, Wvh, n8);
        cvt_f32_to_f16<<<(n8 + 255) / 256, 256, 0, stream>>>(wo, Woh, n8);
    }

    // GEMM: waves = (M/32)*(N/64) = 2048; 8 waves / 256-thread block
    const int gemmBlocks = ((MROWS / 32) * (D_MODEL / 64)) / 8;   // 256

    // Q, K, V projections (V stored transposed per head for the PV WMMA)
    gemm_wmma_f16<<<gemmBlocks, 256, 0, stream>>>(Xh, Wqh, Qh,  nullptr,
                                                  MROWS, D_MODEL, D_MODEL, 0);
    gemm_wmma_f16<<<gemmBlocks, 256, 0, stream>>>(Xh, Wkh, Kh,  nullptr,
                                                  MROWS, D_MODEL, D_MODEL, 0);
    gemm_wmma_f16<<<gemmBlocks, 256, 0, stream>>>(Xh, Wvh, Vth, nullptr,
                                                  MROWS, D_MODEL, D_MODEL, 1);

    // Flash attention: 32 bh * 64 q-tiles = 2048 waves; 4 waves / block
    attn_flash_wmma<<<(BATCH * HEADS * (SEQ / 32)) / 4, 128, 0, stream>>>(
        Qh, Kh, Vth, Ctxh);

    // Output projection -> fp32 d_out
    gemm_wmma_f16<<<gemmBlocks, 256, 0, stream>>>(Ctxh, Woh, nullptr, out,
                                                  MROWS, D_MODEL, D_MODEL, 3);
}
